// LinearAttention_45561013076151
// MI455X (gfx1250) — matrix operand order repaired and checked by reading, not compiled, not run on hardware
//
#include <hip/hip_runtime.h>
#include <stdint.h>

// Problem constants (match reference)
#define B_   8
#define C_   512
#define T_   4096
#define H_   8
#define DH_  64
#define EPS_ 1e-6f

typedef __attribute__((ext_vector_type(16))) _Float16 v16h;
typedef __attribute__((ext_vector_type(8)))  _Float16 v8h;
typedef __attribute__((ext_vector_type(8)))  float    v8f;

typedef unsigned int u32x4 __attribute__((ext_vector_type(4)));
typedef int          i32x8 __attribute__((ext_vector_type(8)));
typedef int          i32x4 __attribute__((ext_vector_type(4)));

// ---------------------------------------------------------------------------
// WMMA fragment loaders (wave32, V_WMMA_F32_16X16X32_F16)
//
// A (16x32 f16, ISA 7.12.2): lane m=lane&15 holds row M=m.
//   half = lane>>4 : VGPR0-7 hold K = half*16 + 0..15 here (contiguous, matches B; ISA table order differs)
// => two 16B loads per lane from a row-major source (ld = K stride in elems).
__device__ __forceinline__ v16h frag_a_rowmajor(const _Float16* base, int ld) {
  const int lane = threadIdx.x & 31;
  const int m    = lane & 15;
  const int half = lane >> 4;
  const _Float16* p = base + (size_t)m * ld + half * 16;
  v8h lo = *(const v8h*)(p);
  v8h hi = *(const v8h*)(p + 8);
  v16h r;
#pragma unroll
  for (int i = 0; i < 8; ++i) { r[i] = lo[i]; r[i + 8] = hi[i]; }
  return r;
}

// B (32x16 f16, same contiguous k order as A, not the ISA table order):
//   lanes 0-15 hold column N=lane with K=0..15, lanes 16-31 hold K=16..31.
// Source must be "N rows, contiguous K" (row n holds k-strip), ld = K stride.
// => one contiguous 32B load per lane.
__device__ __forceinline__ v16h frag_b_nrows(const _Float16* base, int ld) {
  const int lane = threadIdx.x & 31;
  const int n    = lane & 15;
  const int half = lane >> 4;
  return *(const v16h*)(base + (size_t)n * ld + half * 16);
}

__device__ __forceinline__ v8f wmma_f16(v16h a, v16h b, v8f c) {
  return __builtin_amdgcn_wmma_f32_16x16x32_f16(false, a, false, b, (short)0, c,
                                                false, false);
}

// ---------------------------------------------------------------------------
// Tensor Data Mover: async 2D tile load global -> LDS.
// D# bitfields per CDNA5 ISA 8.3 (group0) / 8.4 (group1); groups 2/3 zero (2D).
// esize_code: 0=1B, 1=2B, 2=4B. Dims/strides in element units.
// NOTE: this toolchain's builtin takes 6 args (clang-23 / therock-10.0 lane).
__device__ __forceinline__ void tdm_load_2d(uint32_t lds_off, const void* gptr,
                                            uint32_t esize_code,
                                            uint32_t tile_d0, uint32_t tile_d1,
                                            uint32_t tensor_d0, uint32_t tensor_d1,
                                            uint32_t stride_d0) {
  const uint64_t ga = (uint64_t)(uintptr_t)gptr;
  u32x4 g0;
  g0[0] = 1u;                                           // count=1 (valid), user mode
  g0[1] = lds_off;                                      // lds_addr (bytes)
  g0[2] = (uint32_t)ga;                                 // global_addr[31:0]
  g0[3] = (uint32_t)((ga >> 32) & 0x01FFFFFFu)          // global_addr[56:32]
          | 0x80000000u;                                // type=2 ("image")
  i32x8 g1;
  g1[0] = (int)(esize_code << 16);                      // wg_mask=0, data_size
  g1[1] = (int)((tensor_d0 & 0xFFFFu) << 16);           // tensor_dim0[15:0]
  g1[2] = (int)((tensor_d0 >> 16) | ((tensor_d1 & 0xFFFFu) << 16));
  g1[3] = (int)((tensor_d1 >> 16) | (tile_d0 << 16));   // tile_dim0
  g1[4] = (int)(tile_d1 & 0xFFFFu);                     // tile_dim1 (tile_dim2=0)
  g1[5] = (int)stride_d0;                               // tensor_dim0_stride[31:0]
  g1[6] = 0;                                            // stride hi / dim1_stride lo
  g1[7] = 0;
  const i32x4 z4 = {0, 0, 0, 0};
  const i32x8 z8 = {0, 0, 0, 0, 0, 0, 0, 0};
  __builtin_amdgcn_tensor_load_to_lds(g0, g1, z4, z4, z8, 0);
}

// ---------------------------------------------------------------------------
// Kernel 0: f32 -> f16 weight conversion
__global__ __launch_bounds__(256) void cvt_f32_f16(const float* __restrict__ src,
                                                   _Float16* __restrict__ dst,
                                                   int n) {
  int i = blockIdx.x * 256 + threadIdx.x;
  if (i < n) dst[i] = (_Float16)src[i];
}

// ---------------------------------------------------------------------------
// Kernel 1/5: fused 512x512 conv1x1 GEMM + RMSNorm + per-mode epilogue.
// Grid: (T/16, B). Block: 256 (8 waves). Wave w owns output rows [64w, 64w+64).
// X slab (all 512 input channels x 16 columns) is TDM-loaded to LDS once,
// transposed to f16 [n][k]; the K=512 WMMA loop then runs with no barriers.
enum { MODE_Q = 0, MODE_K = 1, MODE_V = 2, MODE_P = 3 };

template <int MODE, bool SRC_F16>
__global__ __launch_bounds__(256) void feature_gemm(
    const _Float16* __restrict__ W16,   // [C][C] f16 (row = out channel)
    const void*     __restrict__ xsrc,  // [B][C][T] f32 (or f16 if SRC_F16)
    const float*    __restrict__ bias,  // [C]
    const float*    __restrict__ gain,  // [C]
    const uint8_t*  __restrict__ mask,  // [B][T]   (MODE_K)
    const float*    __restrict__ resid, // [B][C][T](MODE_P)
    float*          __restrict__ out32, // MODE_K: Kraw ; MODE_P: final out
    _Float16*       __restrict__ out16) // MODE_Q: Qf   ; MODE_V: Vf
{
  __shared__ float    ldsRaw[C_ * 16];  // 32KB: raw X slab, reused as Y tile
  __shared__ _Float16 ldsBT[16 * C_];   // 16KB: f16 B slab, [n][k] (k contiguous)

  const int b    = blockIdx.y;
  const int t0   = blockIdx.x * 16;
  const int tid  = threadIdx.x;
  const int wave = tid >> 5;
  const int lane = tid & 31;
  const int r0   = wave * 64;

  // --- TDM slab load: X[b][0..C)[t0..t0+16) -> ldsRaw (row k contiguous, 16 wide)
  if (tid < 32) {
    const uint32_t ldsoff = (uint32_t)(size_t)(void*)ldsRaw;
    if constexpr (SRC_F16) {
      tdm_load_2d(ldsoff, (const _Float16*)xsrc + ((size_t)b * C_) * T_ + t0,
                  1u, 16u, C_, T_, C_, T_);
    } else {
      tdm_load_2d(ldsoff, (const float*)xsrc + ((size_t)b * C_) * T_ + t0,
                  2u, 16u, C_, T_, C_, T_);
    }
    __builtin_amdgcn_s_wait_tensorcnt((short)0);
  }
  __syncthreads();

  // --- transpose + convert: ldsRaw[k][n] -> ldsBT[n][k] f16
  for (int e = tid; e < C_ * 16; e += 256) {
    const int k = e >> 4, n = e & 15;
    _Float16 xv;
    if constexpr (SRC_F16) xv = ((const _Float16*)ldsRaw)[e];
    else                   xv = (_Float16)ldsRaw[e];
    ldsBT[n * C_ + k] = xv;
  }
  __syncthreads();

  // --- barrier-free WMMA main loop over K = 512
  const v8f vzero = {0.f, 0.f, 0.f, 0.f, 0.f, 0.f, 0.f, 0.f};
  v8f acc[4] = {vzero, vzero, vzero, vzero};

  for (int kk = 0; kk < C_; kk += 32) {
    const v16h bf = frag_b_nrows(ldsBT + kk, C_);
#pragma unroll
    for (int mi = 0; mi < 4; ++mi) {
      const v16h af =
          frag_a_rowmajor(W16 + (size_t)(r0 + mi * 16) * C_ + kk, C_);
      acc[mi] = wmma_f16(af, bf, acc[mi]);
    }
  }

  // --- spill accumulators (+bias) to LDS (reuse ldsRaw as [row][col] f32)
  float* ldsY = ldsRaw;
#pragma unroll
  for (int mi = 0; mi < 4; ++mi) {
#pragma unroll
    for (int r = 0; r < 8; ++r) {
      const int row = r0 + mi * 16 + r + ((lane >> 4) << 3);
      const int col = lane & 15;
      ldsY[row * 16 + col] = acc[mi][r] + bias[row];
    }
  }
  __syncthreads();

  // --- epilogue: wave handles 2 columns; lane handles rows [16*lane, +16)
  for (int ci = 0; ci < 2; ++ci) {
    const int col = wave * 2 + ci;
    const int t   = t0 + col;

    float vals[16];
    float ss = 0.f;
#pragma unroll
    for (int j = 0; j < 16; ++j) {
      const int row = lane * 16 + j;
      float v = ldsY[row * 16 + col];
      if constexpr (MODE == MODE_P)
        v += resid[((size_t)b * C_ + row) * T_ + t];
      vals[j] = v;
      ss += v * v;
    }
#pragma unroll
    for (int m = 1; m < 32; m <<= 1) ss += __shfl_xor(ss, m, 32);
    const float rstd = rsqrtf(ss * (1.f / C_) + EPS_);

    if constexpr (MODE == MODE_P) {
#pragma unroll
      for (int j = 0; j < 16; ++j) {
        const int row = lane * 16 + j;
        out32[((size_t)b * C_ + row) * T_ + t] = vals[j] * rstd * gain[row];
      }
    } else {
      float f[16];
#pragma unroll
      for (int j = 0; j < 16; ++j) {
        const int row = lane * 16 + j;
        const float y = vals[j] * rstd * gain[row];
        f[j] = (y > 0.f) ? (y + 1.f) : __expf(y);  // elu(y)+1
      }
      if constexpr (MODE == MODE_Q) {
        // softmax over head dim (64 rows = this lane's 16 rows x 4 lanes)
        float mx = f[0];
#pragma unroll
        for (int j = 1; j < 16; ++j) mx = fmaxf(mx, f[j]);
        mx = fmaxf(mx, __shfl_xor(mx, 1, 32));
        mx = fmaxf(mx, __shfl_xor(mx, 2, 32));
        float se = 0.f;
#pragma unroll
        for (int j = 0; j < 16; ++j) { f[j] = __expf(f[j] - mx); se += f[j]; }
        se += __shfl_xor(se, 1, 32);
        se += __shfl_xor(se, 2, 32);
        const float scl = 0.35355339059327373f / se;  // H^-0.5 / sum
#pragma unroll
        for (int j = 0; j < 16; ++j) {
          const int row = lane * 16 + j;
          out16[((size_t)b * C_ + row) * T_ + t] = (_Float16)(f[j] * scl);
        }
      } else if constexpr (MODE == MODE_K) {
        const bool pad = mask[(size_t)b * T_ + t] != 0;
#pragma unroll
        for (int j = 0; j < 16; ++j) {
          const int row = lane * 16 + j;
          out32[((size_t)b * C_ + row) * T_ + t] = pad ? -10000.f : f[j];
        }
      } else {  // MODE_V
#pragma unroll
        for (int j = 0; j < 16; ++j) {
          const int row = lane * 16 + j;
          out16[((size_t)b * C_ + row) * T_ + t] =
              (_Float16)(f[j] * (1.f / DH_));
        }
      }
    }
  }
}

// ---------------------------------------------------------------------------
// Kernel 2: softmax over T per (b,c) row. Writes exp(x-max) f16; sum separate.
__global__ __launch_bounds__(256) void k_time_softmax(
    const float* __restrict__ kraw,  // [B*C][T]
    _Float16*    __restrict__ kf,    // [B*C][T]
    float*       __restrict__ rowsum)// [B*C]
{
  __shared__ float red[8];
  const int row = blockIdx.x;
  const int tid = threadIdx.x;
  const float* src = kraw + (size_t)row * T_;

  float mx = -3.4e38f;
  for (int t = tid; t < T_; t += 256) mx = fmaxf(mx, src[t]);
#pragma unroll
  for (int m = 1; m < 32; m <<= 1) mx = fmaxf(mx, __shfl_xor(mx, m, 32));
  if ((tid & 31) == 0) red[tid >> 5] = mx;
  __syncthreads();
  mx = red[tid & 7];
#pragma unroll
  for (int m = 1; m < 8; m <<= 1) mx = fmaxf(mx, __shfl_xor(mx, m, 32));
  __syncthreads();

  float se = 0.f;
  for (int t = tid; t < T_; t += 256) {
    const float e = __expf(src[t] - mx);
    kf[(size_t)row * T_ + t] = (_Float16)e;
    se += e;
  }
#pragma unroll
  for (int m = 1; m < 32; m <<= 1) se += __shfl_xor(se, m, 32);
  if ((tid & 31) == 0) red[tid >> 5] = se;
  __syncthreads();
  se = red[tid & 7];
#pragma unroll
  for (int m = 1; m < 8; m <<= 1) se += __shfl_xor(se, m, 32);
  if (tid == 0) rowsum[row] = se;
}

// ---------------------------------------------------------------------------
// Kernel 3: ctx[d][e] = sum_t k[d,t]*v[e,t] / rowsum[d]; store transposed f16.
// Grid: B*H blocks, 256 threads (8 waves; wave -> (m = w>>1, n0 = (w&1)*2)).
__global__ __launch_bounds__(256) void ctx_gemm(
    const _Float16* __restrict__ kf,    // [B*C][T]
    const _Float16* __restrict__ vf,    // [B*C][T]
    const float*    __restrict__ rowsum,// [B*C]
    _Float16*       __restrict__ ctxT)  // [B*H][64(e)][64(d)]
{
  const int bh   = blockIdx.x;
  const int bc0  = bh * DH_;  // == b*C + h*64
  const int wave = threadIdx.x >> 5;
  const int lane = threadIdx.x & 31;
  const int m    = wave >> 1;
  const int n0   = (wave & 1) * 2;

  v8f acc0 = {0.f, 0.f, 0.f, 0.f, 0.f, 0.f, 0.f, 0.f};
  v8f acc1 = acc0;
  const _Float16* abase = kf + ((size_t)bc0 + m * 16) * T_;
  const _Float16* b0    = vf + ((size_t)bc0 + n0 * 16) * T_;
  const _Float16* b1    = vf + ((size_t)bc0 + (n0 + 1) * 16) * T_;

  for (int t = 0; t < T_; t += 32) {
    const v16h af  = frag_a_rowmajor(abase + t, T_);
    const v16h bf0 = frag_b_nrows(b0 + t, T_);
    const v16h bf1 = frag_b_nrows(b1 + t, T_);
    acc0 = wmma_f16(af, bf0, acc0);
    acc1 = wmma_f16(af, bf1, acc1);
  }

#pragma unroll
  for (int r = 0; r < 8; ++r) {
    const int d     = m * 16 + r + ((lane >> 4) << 3);
    const float inv = 1.f / rowsum[bc0 + d];
    const int e0    = n0 * 16 + (lane & 15);
    ctxT[(size_t)bh * 4096 + (size_t)e0 * 64 + d]        = (_Float16)(acc0[r] * inv);
    ctxT[(size_t)bh * 4096 + (size_t)(e0 + 16) * 64 + d] = (_Float16)(acc1[r] * inv);
  }
}

// ---------------------------------------------------------------------------
// Kernel 4: attn[e][t] = sum_d ctxT[e][d] * q[d][t]  (K=64).
// Grid: (T/128, B*H), 256 threads. Wave w owns 16 cols, all 64 e-rows.
// Q tile (64 x 128, row stride T) is TDM-loaded, then transposed in LDS.
__global__ __launch_bounds__(256) void attn_gemm(
    const _Float16* __restrict__ ctxT,  // [B*H][64(e)][64(d)]
    const _Float16* __restrict__ qf,    // [B*C][T]
    _Float16*       __restrict__ attn)  // [B*C][T]
{
  __shared__ _Float16 ldsQraw[DH_ * 128];  // [d][t] as loaded by TDM
  __shared__ _Float16 ldsQ[128 * DH_];     // transposed [t][d]
  const int bh   = blockIdx.y;
  const int t0   = blockIdx.x * 128;
  const int bc0  = bh * DH_;
  const int tid  = threadIdx.x;
  const int wave = tid >> 5;
  const int lane = tid & 31;

  if (tid < 32) {
    tdm_load_2d((uint32_t)(size_t)(void*)ldsQraw,
                qf + (size_t)bc0 * T_ + t0,
                1u, 128u, DH_, T_, DH_, T_);
    __builtin_amdgcn_s_wait_tensorcnt((short)0);
  }
  __syncthreads();

  for (int e = tid; e < DH_ * 128; e += 256) {
    const int d = e >> 7, t = e & 127;
    ldsQ[t * DH_ + d] = ldsQraw[e];
  }
  __syncthreads();

  const v8f vzero = {0.f, 0.f, 0.f, 0.f, 0.f, 0.f, 0.f, 0.f};
  v8f acc[4] = {vzero, vzero, vzero, vzero};

#pragma unroll
  for (int kb = 0; kb < 64; kb += 32) {
    const int n    = lane & 15;
    const int half = lane >> 4;
    const v16h bf =
        *(const v16h*)(&ldsQ[(wave * 16 + n) * 64 + kb + half * 16]);
#pragma unroll
    for (int mi = 0; mi < 4; ++mi) {
      const v16h af = frag_a_rowmajor(
          ctxT + (size_t)bh * 4096 + (size_t)(mi * 16) * 64 + kb, 64);
      acc[mi] = wmma_f16(af, bf, acc[mi]);
    }
  }

#pragma unroll
  for (int mi = 0; mi < 4; ++mi) {
#pragma unroll
    for (int r = 0; r < 8; ++r) {
      const int e = mi * 16 + r + ((lane >> 4) << 3);
      const int t = t0 + wave * 16 + (lane & 15);
      attn[((size_t)bc0 + e) * T_ + t] = (_Float16)acc[mi][r];
    }
  }
}

// ---------------------------------------------------------------------------
extern "C" void kernel_launch(void* const* d_in, const int* in_sizes, int n_in,
                              void* d_out, int out_size, void* d_ws,
                              size_t ws_size, hipStream_t stream) {
  (void)in_sizes; (void)n_in; (void)out_size; (void)ws_size;
  const float*   q_in  = (const float*)d_in[0];
  const float*   k_in  = (const float*)d_in[1];
  const float*   v_in  = (const float*)d_in[2];
  const uint8_t* mask  = (const uint8_t*)d_in[3];
  const float* Wq = (const float*)d_in[4];
  const float* bq = (const float*)d_in[5];
  const float* gq = (const float*)d_in[6];
  const float* Wk = (const float*)d_in[7];
  const float* bk = (const float*)d_in[8];
  const float* gk = (const float*)d_in[9];
  const float* Wv = (const float*)d_in[10];
  const float* bv = (const float*)d_in[11];
  const float* gv = (const float*)d_in[12];
  const float* Wp = (const float*)d_in[13];
  const float* bp = (const float*)d_in[14];
  const float* g_out = (const float*)d_in[15];
  float* out = (float*)d_out;

  char* ws = (char*)d_ws;
  size_t off = 0;
  auto alloc = [&](size_t bytes) -> void* {
    void* p = (void*)(ws + off);
    off += (bytes + 255) & ~(size_t)255;
    return p;
  };
  const size_t nW = (size_t)C_ * C_;       // 262144
  const size_t nX = (size_t)B_ * C_ * T_;  // 16777216
  _Float16* W16q = (_Float16*)alloc(nW * 2);
  _Float16* W16k = (_Float16*)alloc(nW * 2);
  _Float16* W16v = (_Float16*)alloc(nW * 2);
  _Float16* W16p = (_Float16*)alloc(nW * 2);
  _Float16* Qf   = (_Float16*)alloc(nX * 2);
  _Float16* Vf   = (_Float16*)alloc(nX * 2);
  float*    Kraw = (float*)   alloc(nX * 4);
  _Float16* Kf   = (_Float16*)alloc(nX * 2);
  float*    rsum = (float*)   alloc((size_t)B_ * C_ * 4);
  _Float16* ctxT = (_Float16*)alloc((size_t)B_ * H_ * DH_ * DH_ * 2);
  _Float16* attn = (_Float16*)alloc(nX * 2);

  // 0) weight conversion (f32 -> f16)
  cvt_f32_f16<<<(int)(nW / 256), 256, 0, stream>>>(Wq, W16q, (int)nW);
  cvt_f32_f16<<<(int)(nW / 256), 256, 0, stream>>>(Wk, W16k, (int)nW);
  cvt_f32_f16<<<(int)(nW / 256), 256, 0, stream>>>(Wv, W16v, (int)nW);
  cvt_f32_f16<<<(int)(nW / 256), 256, 0, stream>>>(Wp, W16p, (int)nW);

  // 1) fused conv1x1 + RMSNorm + elu+1 (+ per-mode epilogues)
  dim3 g1(T_ / 16, B_);
  feature_gemm<MODE_Q, false><<<g1, 256, 0, stream>>>(
      W16q, q_in, bq, gq, nullptr, nullptr, nullptr, Qf);
  feature_gemm<MODE_K, false><<<g1, 256, 0, stream>>>(
      W16k, k_in, bk, gk, mask, nullptr, Kraw, nullptr);
  feature_gemm<MODE_V, false><<<g1, 256, 0, stream>>>(
      W16v, v_in, bv, gv, nullptr, nullptr, nullptr, Vf);

  // 2) key softmax over time
  k_time_softmax<<<B_ * C_, 256, 0, stream>>>(Kraw, Kf, rsum);

  // 3) ctx = k @ v^T per head (K = T)
  ctx_gemm<<<B_ * H_, 256, 0, stream>>>(Kf, Vf, rsum, ctxT);

  // 4) out = ctx^T @ q per head (K = 64)
  attn_gemm<<<dim3(T_ / 128, B_ * H_), 256, 0, stream>>>(ctxT, Qf, attn);

  // 5) final projection + residual + RMSNorm
  feature_gemm<MODE_P, true><<<g1, 256, 0, stream>>>(
      W16p, attn, bp, g_out, nullptr, q_in, out, nullptr);
}
